// Edge_Regressor_1099511627811
// MI455X (gfx1250) — matrix operand order repaired and checked by reading, not compiled, not run on hardware
//
#include <hip/hip_runtime.h>
#include <hip/hip_bf16.h>
#include <stdint.h>

// ---------------------------------------------------------------------------
// Edge regressor, fused MLP on MI455X (gfx1250, wave32, WMMA bf16)
// B=4, N=256, D=128, E=64, H=256.
// layer1 = relu(P[i] + edges@W1b + Q[j])   (P includes b1; P/Q precomputed)
// layer2 = relu(A1@W2 + b2);  out = A2@W3 + b3
// Persistent workgroups: 512 WGs, each stages bf16 W2T/W1bT into LDS ONCE
// via global_load_async_to_lds_b128 (ASYNCcnt), then loops over 8 row-tiles.
// ---------------------------------------------------------------------------

typedef __attribute__((ext_vector_type(16))) __bf16 v16bf;
typedef __attribute__((ext_vector_type(8)))  float  v8f;

#define ROWS 64                 // pairs per tile (all share b and i)
#define GRID_MAIN 512           // persistent WGs
#define NUM_TILES 4096          // (B*N*N)/ROWS
#define TILES_PER_WG (NUM_TILES / GRID_MAIN)   // 8

// LDS strides in bf16 elements (multiples of 8 -> 16B-aligned frag loads,
// padded to break the 512B bank stride)
#define W2T_STRIDE 264
#define W1B_STRIDE 72
#define A1_STRIDE  264
#define E_STRIDE   72

#define OFF_W2T 0
#define SZ_W2T  (256 * W2T_STRIDE * 2)          // 135168 B = 8448 x 16B
#define OFF_W1B (OFF_W2T + SZ_W2T)
#define SZ_W1B  (256 * W1B_STRIDE * 2)          // 36864 B = 2304 x 16B
#define OFF_A1  (OFF_W1B + SZ_W1B)
#define SZ_A1   (ROWS * A1_STRIDE * 2)          // 33792
#define OFF_E   (OFF_A1 + SZ_A1)
#define SZ_E    (ROWS * E_STRIDE * 2)           // 9216
#define OFF_RS  (OFF_E + SZ_E)
#define SZ_RS   (ROWS * 4)
#define SMEM_TOTAL (OFF_RS + SZ_RS)             // 215296 B < 320KB LDS/WGP

#define W2_XFERS_PER_THREAD  (SZ_W2T / 16 / 256)   // 33
#define W1B_XFERS_PER_THREAD (SZ_W1B / 16 / 256)   // 9

union Frag { uint4 q[2]; v16bf v; };

__device__ __forceinline__ uint32_t bfbits(float x) {          // RNE fp32->bf16
    union { float f; uint32_t u; } v; v.f = x;
    return (v.u + 0x7fffu + ((v.u >> 16) & 1u)) >> 16;
}
__device__ __forceinline__ __bf16 f2bf(float x) {
    union { uint16_t s; __bf16 b; } o; o.s = (uint16_t)bfbits(x);
    return o.b;
}

// Async DMA: global (saddr + 32b voffset) -> LDS (per-lane 32b address).
// GLOBAL_LOAD_ASYNC_TO_LDS_B128, tracked by ASYNCcnt.
__device__ __forceinline__ void async_g2l_b128(uint32_t lds_addr,
                                               uint32_t g_off,
                                               const void* g_base) {
    asm volatile("global_load_async_to_lds_b128 %0, %1, %2"
                 :: "v"(lds_addr), "v"(g_off), "s"(g_base) : "memory");
}

// ---------------------------------------------------------------------------
// Prep 1: P[bn,h] = b1[h] + sum_d nodes[bn,d]*W1[d,h]
//         Q[bn,h] =         sum_d nodes[bn,d]*W1[192+d,h]
// ---------------------------------------------------------------------------
extern "C" __global__ __launch_bounds__(256)
void pq_kernel(const float* __restrict__ nodes,
               const float* __restrict__ W1,
               const float* __restrict__ b1,
               float* __restrict__ P,
               float* __restrict__ Q)
{
    __shared__ float snode[128];
    const int bn = blockIdx.x;            // b*256 + n
    const int h  = threadIdx.x;           // 0..255
    if (h < 128) snode[h] = nodes[(size_t)bn * 128 + h];
    __syncthreads();
    float p = b1[h], q = 0.0f;
    #pragma unroll 8
    for (int d = 0; d < 128; ++d) {
        const float nv = snode[d];
        p = fmaf(nv, W1[d * 256 + h], p);
        q = fmaf(nv, W1[(192 + d) * 256 + h], q);
    }
    P[(size_t)bn * 256 + h] = p;
    Q[(size_t)bn * 256 + h] = q;
}

// ---------------------------------------------------------------------------
// Prep 2: one-time bf16 transpose of W2 and W1b into d_ws with the padded
// LDS strides baked in, so the main kernel staging is a flat b128 copy.
// ---------------------------------------------------------------------------
extern "C" __global__ __launch_bounds__(256)
void wconv_kernel(const float* __restrict__ W1,
                  const float* __restrict__ W2,
                  uint16_t* __restrict__ wsW2T,
                  uint16_t* __restrict__ wsW1bT)
{
    const int k = blockIdx.x;             // 0..255
    const int c = threadIdx.x;            // 0..255
    wsW2T[c * W2T_STRIDE + k] = (uint16_t)bfbits(W2[k * 256 + c]);
    if (k < 64)
        wsW1bT[c * W1B_STRIDE + k] = (uint16_t)bfbits(W1[(128 + k) * 256 + c]);
}

// ---------------------------------------------------------------------------
// Main: fused edge MLP. Persistent WG (8 wave32) processes 8 tiles of 64
// pairs each; weights live in LDS for the whole kernel.
// ---------------------------------------------------------------------------
extern "C" __global__ __launch_bounds__(256)
void edge_mlp_kernel(const float* __restrict__ edges,
                     const float* __restrict__ b2,
                     const float* __restrict__ W3,
                     const float* __restrict__ b3,
                     const float* __restrict__ P,      // [B*N, 256] (has b1)
                     const float* __restrict__ Q,      // [B*N, 256]
                     const uint16_t* __restrict__ gW2T,  // pre-bf16, padded
                     const uint16_t* __restrict__ gW1bT, // pre-bf16, padded
                     float* __restrict__ out)
{
    extern __shared__ char smem[];
    __bf16* sW2  = (__bf16*)(smem + OFF_W2T);  // [col][k]  W2 transposed
    __bf16* sW1b = (__bf16*)(smem + OFF_W1B);  // [col][kk] W1b transposed
    __bf16* sA1  = (__bf16*)(smem + OFF_A1);   // [row][k]  layer-1 act
    __bf16* sE   = (__bf16*)(smem + OFF_E);    // [row][e]  edges tile
    float*  sRS  = (float*)(smem + OFF_RS);    // [row]     layer-3 rowsum
    uint2*  sE2  = (uint2*)sE;                 // 18 uint2 per padded row

    const int tid = threadIdx.x;

    // ---- once per WG: async weight DMA, W1b first (needed first), then W2 --
    const uint32_t ldsW1b = (uint32_t)(uintptr_t)sW1b;
    const uint32_t ldsW2  = (uint32_t)(uintptr_t)sW2;
    #pragma unroll
    for (int t = 0; t < W1B_XFERS_PER_THREAD; ++t) {          // 9 x 512B/wave
        const uint32_t off = (uint32_t)(tid + t * 256) * 16u;
        async_g2l_b128(ldsW1b + off, off, gW1bT);
    }
    #pragma unroll 4
    for (int t = 0; t < W2_XFERS_PER_THREAD; ++t) {           // 33 x 512B/wave
        const uint32_t off = (uint32_t)(tid + t * 256) * 16u;
        async_g2l_b128(ldsW2 + off, off, gW2T);
    }

    // ---- wave tiling: 8 waves in 2x4, each owns 32 rows x 64 cols ----
    const int lane = tid & 31;
    const int wv   = tid >> 5;
    const int rw   = wv >> 2;          // 0..1
    const int cw   = wv & 3;           // 0..3
    const int g    = lane >> 4;        // half-wave group
    const int lr   = lane & 15;
    const int rowbase = rw * 32;
    const int colbase = cw * 64;

    // per-lane constants, hoisted out of the tile loop
    float b2c[4], w3c[4];
    #pragma unroll
    for (int ct = 0; ct < 4; ++ct) {
        const int col = colbase + ct*16 + lr;
        b2c[ct] = b2[col];
        w3c[ct] = W3[col];
    }
    const float b3v = b3[0];

    // ---- prefetch first tile's edges into registers ----
    int tile = blockIdx.x;
    float4 pf[4];
    #pragma unroll
    for (int t = 0; t < 4; ++t) {
        const int base = (tid + t * 256) * 4;  // element index, 4-aligned
        const int r = base >> 6, e = base & 63;
        pf[t] = *(const float4*)(edges + ((size_t)tile * ROWS + r) * 64 + e);
    }

    // W1b portion (first 9 in-order async ops) must be done before layer 1.
    asm volatile("s_wait_asynccnt 33" ::: "memory");

    #pragma unroll 1
    for (int it = 0; it < TILES_PER_WG; ++it, tile += GRID_MAIN) {
        const int p0 = tile * ROWS;            // flat pair index incl. batch
        const int b  = p0 >> 16;
        const int i  = (p0 >> 8) & 255;
        const int j0 = p0 & 255;

        // ---- stage edges tile (prefetched fp32 -> bf16), zero rowsum ----
        #pragma unroll
        for (int t = 0; t < 4; ++t) {
            const int base = (tid + t * 256) * 4;
            const int r = base >> 6, e = base & 63;
            uint2 pk;
            pk.x = bfbits(pf[t].x) | (bfbits(pf[t].y) << 16);
            pk.y = bfbits(pf[t].z) | (bfbits(pf[t].w) << 16);
            sE2[r * 18 + (e >> 2)] = pk;
        }
        if (tid < ROWS) sRS[tid] = 0.0f;
        __syncthreads();

        v8f acc[2][4];
        #pragma unroll
        for (int rt = 0; rt < 2; ++rt)
            #pragma unroll
            for (int ct = 0; ct < 4; ++ct)
                acc[rt][ct] = (v8f){0,0,0,0,0,0,0,0};

        // ====== layer 1: E[64x64] @ W1b[64x256] (W2 DMA overlaps, tile 0) ===
        #pragma unroll
        for (int kb = 0; kb < 64; kb += 32) {
            Frag fa[2];
            #pragma unroll
            for (int rt = 0; rt < 2; ++rt) {
                const __bf16* ap = sE + (rowbase + rt*16 + lr) * E_STRIDE + kb + g*8;
                fa[rt].q[0] = *(const uint4*)ap;          // K = kb+g*8    ..+7
                fa[rt].q[1] = *(const uint4*)(ap + 16);   // K = kb+16+g*8 ..+7
            }
            #pragma unroll
            for (int ct = 0; ct < 4; ++ct) {
                Frag fb;
                const __bf16* bp = sW1b + (colbase + ct*16 + lr) * W1B_STRIDE + kb + g*8;
                fb.q[0] = *(const uint4*)bp;              // K = kb+g*8    ..+7
                fb.q[1] = *(const uint4*)(bp + 16);       // K = kb+16+g*8 ..+7
                #pragma unroll
                for (int rt = 0; rt < 2; ++rt)
                    acc[rt][ct] = __builtin_amdgcn_wmma_f32_16x16x32_bf16(
                        false, fa[rt].v, false, fb.v, (short)0, acc[rt][ct],
                        false, false);
            }
        }

        // ---- epilogue 1: + P[i] + Q[j], relu, store bf16 A1 to LDS ----
        const float* Prow  = P + (size_t)(b * 256 + i)  * 256;
        const float* Qbase = Q + (size_t)(b * 256 + j0) * 256;
        #pragma unroll
        for (int ct = 0; ct < 4; ++ct) {
            const int col = colbase + ct*16 + lr;        // C-layout: N = lane%16
            const float pv = Prow[col];
            #pragma unroll
            for (int rt = 0; rt < 2; ++rt)
                #pragma unroll
                for (int e = 0; e < 8; ++e) {            // C-layout: M = e + 8*g
                    const int r = rowbase + rt*16 + g*8 + e;
                    float v = acc[rt][ct][e] + pv + Qbase[(size_t)r * 256 + col];
                    sA1[r * A1_STRIDE + col] = f2bf(fmaxf(v, 0.0f));
                }
        }
        // W2 DMA must have fully landed before any wave passes this barrier
        // (no-op after tile 0)
        asm volatile("s_wait_asynccnt 0" ::: "memory");
        __syncthreads();

        // ---- prefetch next tile's edges (hides under layer-2 WMMAs) ----
        if (it + 1 < TILES_PER_WG) {
            const size_t np0 = (size_t)(tile + GRID_MAIN) * ROWS;
            #pragma unroll
            for (int t = 0; t < 4; ++t) {
                const int base = (tid + t * 256) * 4;
                const int r = base >> 6, e = base & 63;
                pf[t] = *(const float4*)(edges + (np0 + r) * 64 + e);
            }
        }

        // =============== layer 2: A1[64x256] @ W2[256x256] ===============
        #pragma unroll
        for (int rt = 0; rt < 2; ++rt)
            #pragma unroll
            for (int ct = 0; ct < 4; ++ct)
                acc[rt][ct] = (v8f){0,0,0,0,0,0,0,0};

        #pragma unroll
        for (int kb = 0; kb < 256; kb += 32) {
            Frag fa[2];
            #pragma unroll
            for (int rt = 0; rt < 2; ++rt) {
                const __bf16* ap = sA1 + (rowbase + rt*16 + lr) * A1_STRIDE + kb + g*8;
                fa[rt].q[0] = *(const uint4*)ap;
                fa[rt].q[1] = *(const uint4*)(ap + 16);
            }
            #pragma unroll
            for (int ct = 0; ct < 4; ++ct) {
                Frag fb;
                const __bf16* bp = sW2 + (colbase + ct*16 + lr) * W2T_STRIDE + kb + g*8;
                fb.q[0] = *(const uint4*)bp;
                fb.q[1] = *(const uint4*)(bp + 16);
                #pragma unroll
                for (int rt = 0; rt < 2; ++rt)
                    acc[rt][ct] = __builtin_amdgcn_wmma_f32_16x16x32_bf16(
                        false, fa[rt].v, false, fb.v, (short)0, acc[rt][ct],
                        false, false);
            }
        }

        // ---- epilogue 2 + layer 3: relu(. + b2) dot W3, LDS row reduce ----
        #pragma unroll
        for (int rt = 0; rt < 2; ++rt)
            #pragma unroll
            for (int e = 0; e < 8; ++e) {
                float s = 0.0f;
                #pragma unroll
                for (int ct = 0; ct < 4; ++ct)
                    s = fmaf(fmaxf(acc[rt][ct][e] + b2c[ct], 0.0f), w3c[ct], s);
                atomicAdd(&sRS[rowbase + rt*16 + g*8 + e], s);   // ds_add_f32
            }
        __syncthreads();

        if (tid < ROWS) out[p0 + tid] = sRS[tid] + b3v;
        __syncthreads();   // protect sRS/sE/sA1 reuse in the next tile
    }
}

// ---------------------------------------------------------------------------
extern "C" void kernel_launch(void* const* d_in, const int* in_sizes, int n_in,
                              void* d_out, int out_size, void* d_ws, size_t ws_size,
                              hipStream_t stream) {
    (void)in_sizes; (void)n_in; (void)out_size; (void)ws_size;
    const float* nodes = (const float*)d_in[0];
    const float* edges = (const float*)d_in[1];
    const float* W1    = (const float*)d_in[2];
    const float* b1    = (const float*)d_in[3];
    const float* W2    = (const float*)d_in[4];
    const float* b2    = (const float*)d_in[5];
    const float* W3    = (const float*)d_in[6];
    const float* b3    = (const float*)d_in[7];

    float* P = (float*)d_ws;                       // [1024, 256] fp32
    float* Q = P + 1024 * 256;                     // [1024, 256] fp32
    uint16_t* wsW2T  = (uint16_t*)(Q + 1024 * 256);        // 256*264 bf16
    uint16_t* wsW1bT = wsW2T + 256 * W2T_STRIDE;           // 256*72  bf16
    float* out = (float*)d_out;                    // [B*N*N]

    static bool attr_set = false;                  // idempotent, deterministic
    if (!attr_set) {
        hipFuncSetAttribute((const void*)edge_mlp_kernel,
                            hipFuncAttributeMaxDynamicSharedMemorySize,
                            SMEM_TOTAL);
        attr_set = true;
    }

    pq_kernel<<<1024, 256, 0, stream>>>(nodes, W1, b1, P, Q);
    wconv_kernel<<<256, 256, 0, stream>>>(W1, W2, wsW2T, wsW1bT);
    edge_mlp_kernel<<<GRID_MAIN, 256, SMEM_TOTAL, stream>>>(
        edges, b2, W3, b3, P, Q, wsW2T, wsW1bT, out);
}
